// MultiHeadAttention_4260607558224
// MI455X (gfx1250) — compile-verified, matrix operand order checked by reading, not run on hardware
//
#include <hip/hip_runtime.h>
#include <stddef.h>

// ---------------------------------------------------------------------------
// MHA (B=4, S=2048, D=1024, H=16, HD=64) for MI455X (gfx1250, wave32, WMMA)
// bf16 inputs / fp32 accumulation via v_wmma_f32_16x16x32_bf16.
// ---------------------------------------------------------------------------

#define B_  4
#define S_  2048
#define D_  1024
#define H_  16
#define HD_ 64
#define M_  (B_ * S_)          // 8192 total rows

typedef __attribute__((ext_vector_type(16))) __bf16   v16bf;
typedef __attribute__((ext_vector_type(8)))  float    v8f;
typedef __attribute__((ext_vector_type(4)))  unsigned u32x4;

union Frag {                 // 32 bytes: one 16-bit A/B WMMA fragment
    u32x4 u[2];
    v16bf v;
};
union Pack8 {                // 8 bf16 packed for a single b128 store
    unsigned short s[8];
    u32x4 q;
};

__device__ __forceinline__ unsigned short f2bf(float f) {
    union { float f; unsigned u; } c; c.f = f;
    unsigned r = c.u + 0x7FFFu + ((c.u >> 16) & 1u);   // round-to-nearest-even
    return (unsigned short)(r >> 16);
}

// Load one 16-bit WMMA fragment whose K runs contiguously from `base`.
// ISA layout: lanes 0-15 hold K = 0..7 and 16..23; lanes 16-31 hold 8..15 / 24..31.
__device__ __forceinline__ v16bf load_frag(const unsigned short* base, int grp) {
    Frag f;
    f.u[0] = *(const u32x4*)(base + grp * 8);
    f.u[1] = *(const u32x4*)(base + 16 + grp * 8);
    return f.v;
}

__device__ __forceinline__ v8f wmma_bf16(v16bf a, v16bf b, v8f c) {
    return __builtin_amdgcn_wmma_f32_16x16x32_bf16(
        /*neg_a=*/false, a, /*neg_b=*/false, b,
        /*c_mod=*/(short)0, c, /*reuse_a=*/false, /*reuse_b=*/false);
}

// ---------------------------------------------------------------------------
// Conversion / transpose kernels (fp32 -> bf16)
// ---------------------------------------------------------------------------
__global__ void cvt_x(const float* __restrict__ src, unsigned short* __restrict__ dst, int n) {
    for (int i = blockIdx.x * blockDim.x + threadIdx.x; i < n; i += gridDim.x * blockDim.x)
        dst[i] = f2bf(src[i]);
}

// W: [H, D, HD] -> Wt: [H, HD, D]   (makes B-fragment loads contiguous in K=D)
__global__ void cvt_w_t(const float* __restrict__ src, unsigned short* __restrict__ dst) {
    const int n = H_ * D_ * HD_;
    for (int i = blockIdx.x * blockDim.x + threadIdx.x; i < n; i += gridDim.x * blockDim.x) {
        int h   = i / (D_ * HD_);
        int rem = i - h * (D_ * HD_);
        int d   = rem / HD_;
        int c   = rem - d * HD_;
        dst[(h * HD_ + c) * D_ + d] = f2bf(src[i]);
    }
}

// Wp: [D, D] -> WpT: [n, d]
__global__ void cvt_wp_t(const float* __restrict__ src, unsigned short* __restrict__ dst) {
    const int n = D_ * D_;
    for (int i = blockIdx.x * blockDim.x + threadIdx.x; i < n; i += gridDim.x * blockDim.x) {
        int nr = i / D_;
        int d  = i - nr * D_;
        dst[i] = f2bf(src[d * D_ + nr]);
    }
}

// ---------------------------------------------------------------------------
// Projection GEMM: per head  C[8192, 64] = xb[8192, 1024] @ Wt[h]^T
// grid = (H, M/64), 256 threads = 8 waves; wave (wm,wn) owns a 16x32 C tile.
// transposeOut=0 -> out[bh, s, hd]  (Q, K) ; transposeOut=1 -> out[bh, hd, s] (V)
// ---------------------------------------------------------------------------
__global__ void __launch_bounds__(256)
gemm_proj(const unsigned short* __restrict__ xb,   // [M, D] bf16
          const unsigned short* __restrict__ wt,   // [H, HD, D] bf16
          unsigned short* __restrict__ out,
          int transposeOut) {
    const int h    = blockIdx.x;
    const int tid  = threadIdx.x;
    const int wave = tid >> 5, lane = tid & 31;
    const int grp  = lane >> 4, lr = lane & 15;
    const int wm   = wave >> 1, wn = wave & 1;
    const int m0   = blockIdx.y * 64 + wm * 16;
    const int n0   = wn * 32;

    const unsigned short* arow  = xb + (size_t)(m0 + lr) * D_;
    const unsigned short* b0row = wt + (size_t)h * HD_ * D_ + (size_t)(n0 + lr) * D_;
    const unsigned short* b1row = b0row + (size_t)16 * D_;

    v8f c0 = {}, c1 = {};
    for (int kb = 0; kb < D_; kb += 32) {
        v16bf a  = load_frag(arow  + kb, grp);
        v16bf b0 = load_frag(b0row + kb, grp);
        v16bf b1 = load_frag(b1row + kb, grp);
        c0 = wmma_bf16(a, b0, c0);
        c1 = wmma_bf16(a, b1, c1);
    }

    const int    b  = m0 >> 11;            // / S_
    const int    s0 = m0 & (S_ - 1);
    const size_t bh = (size_t)b * H_ + h;

    if (transposeOut) {
        // V^T layout [bh, hd, s]: 8 consecutive s per lane -> packed b128 store
        for (int nt = 0; nt < 2; ++nt) {
            v8f cc = nt ? c1 : c0;
            Pack8 p;
#pragma unroll
            for (int r = 0; r < 8; ++r) p.s[r] = f2bf(cc[r]);
            const int col = n0 + nt * 16 + lr;
            *(u32x4*)(out + (bh * HD_ + col) * S_ + s0 + grp * 8) = p.q;
        }
    } else {
        for (int nt = 0; nt < 2; ++nt) {
            v8f cc = nt ? c1 : c0;
            const int col = n0 + nt * 16 + lr;
#pragma unroll
            for (int r = 0; r < 8; ++r) {
                const int srow = s0 + grp * 8 + r;
                out[(bh * S_ + srow) * HD_ + col] = f2bf(cc[r]);
            }
        }
    }
}

// ---------------------------------------------------------------------------
// Flash attention (causal, online softmax).
// grid = (B*H, S/64), 128 threads = 4 independent waves, 16 rows each.
// ---------------------------------------------------------------------------
__global__ void __launch_bounds__(128)
flash_attn(const unsigned short* __restrict__ qb,   // [BH, S, HD]
           const unsigned short* __restrict__ kb_,  // [BH, S, HD]
           const unsigned short* __restrict__ vtb,  // [BH, HD, S]
           unsigned short* __restrict__ attn) {     // [B, S, D]
    __shared__ unsigned short pbuf[4][16][40];      // padded: 80B row stride (16B aligned)

    const int bh   = blockIdx.x;
    const int tid  = threadIdx.x;
    const int wave = tid >> 5, lane = tid & 31;
    const int grp  = lane >> 4, lr = lane & 15;
    const int row0 = blockIdx.y * 64 + wave * 16;

    // Q A-fragments (held in registers for the whole KV loop)
    const unsigned short* qrow = qb + ((size_t)bh * S_ + row0 + lr) * HD_;
    const v16bf qa0 = load_frag(qrow + 0,  grp);
    const v16bf qa1 = load_frag(qrow + 32, grp);

    const unsigned short* kbase = kb_ + (size_t)bh * S_ * HD_;
    const unsigned short* vbase = vtb + (size_t)bh * HD_ * S_;

    float mi[8], li[8];
    v8f   o[4] = {v8f{}, v8f{}, v8f{}, v8f{}};
#pragma unroll
    for (int r = 0; r < 8; ++r) { mi[r] = -3.0e38f; li[r] = 0.0f; }

    const int steps = (row0 + 16 + 31) >> 5;        // causal trip count (32 keys/step)
    for (int jt = 0; jt < steps; ++jt) {
        const int j0 = jt << 5;

        // ---- scores S = (Q @ K^T) * rsqrt(HD), 16x32 in two 16x16 fragments
        const unsigned short* k0 = kbase + (size_t)(j0 + lr) * HD_;
        const unsigned short* k1 = k0 + (size_t)16 * HD_;
        v8f s0v = {}, s1v = {};
        s0v = wmma_bf16(qa0, load_frag(k0 + 0,  grp), s0v);
        s0v = wmma_bf16(qa1, load_frag(k0 + 32, grp), s0v);
        s1v = wmma_bf16(qa0, load_frag(k1 + 0,  grp), s1v);
        s1v = wmma_bf16(qa1, load_frag(k1 + 32, grp), s1v);

        if (jt + 1 < steps)                          // global_prefetch next K tile
            __builtin_prefetch(kbase + (size_t)(j0 + 32 + lr) * HD_, 0, 0);

        // ---- causal mask + online softmax (row m lives across one 16-lane half)
        const int rbase = row0 + grp * 8;
        const int key0  = j0 + lr, key1 = j0 + 16 + lr;
#pragma unroll
        for (int r = 0; r < 8; ++r) {
            const int row = rbase + r;
            float x0 = s0v[r] * 0.125f;              // 1/sqrt(64)
            float x1 = s1v[r] * 0.125f;
            if (key0 > row) x0 = -3.0e38f;
            if (key1 > row) x1 = -3.0e38f;

            float mr = fmaxf(x0, x1);
#pragma unroll
            for (int off = 1; off < 16; off <<= 1)
                mr = fmaxf(mr, __shfl_xor(mr, off, 16));
            const float mnew  = fmaxf(mi[r], mr);
            const float scale = __expf(mi[r] - mnew);
            const float p0 = __expf(x0 - mnew);
            const float p1 = __expf(x1 - mnew);
            float rs = p0 + p1;
#pragma unroll
            for (int off = 1; off < 16; off <<= 1)
                rs += __shfl_xor(rs, off, 16);
            li[r] = li[r] * scale + rs;
            mi[r] = mnew;
#pragma unroll
            for (int nt = 0; nt < 4; ++nt) o[nt][r] *= scale;

            const int m = grp * 8 + r;               // C-layout -> LDS row-major P
            pbuf[wave][m][lr]      = f2bf(p0);
            pbuf[wave][m][16 + lr] = f2bf(p1);
        }

        // per-wave LDS ordering: stores complete before the A-fragment reload
        asm volatile("s_wait_dscnt 0" ::: "memory");

        Frag pf;                                     // P as 16x32 A fragment
        pf.u[0] = *(const u32x4*)&pbuf[wave][lr][grp * 8];
        pf.u[1] = *(const u32x4*)&pbuf[wave][lr][16 + grp * 8];

        // ---- O += P @ V   (V^T rows are contiguous in the key dimension)
#pragma unroll
        for (int nt = 0; nt < 4; ++nt) {
            const unsigned short* vrow = vbase + (size_t)(nt * 16 + lr) * S_ + j0;
            o[nt] = wmma_bf16(pf.v, load_frag(vrow, grp), o[nt]);
        }
    }

    // ---- normalize and write concat-head output [B, S, D] (bf16)
    const int b = bh >> 4, h = bh & 15;
#pragma unroll
    for (int nt = 0; nt < 4; ++nt) {
        const int col = h * HD_ + nt * 16 + lr;
#pragma unroll
        for (int r = 0; r < 8; ++r) {
            const int row = row0 + grp * 8 + r;
            attn[((size_t)b * S_ + row) * D_ + col] = f2bf(o[nt][r] / li[r]);
        }
    }
}

// ---------------------------------------------------------------------------
// Output projection: out[8192,1024] = attn @ Wp + bp   (fp32 out)
// grid = (M/64, D/128), 256 threads = 8 waves; wave owns 16x64.
// ---------------------------------------------------------------------------
__global__ void __launch_bounds__(256)
gemm_out(const unsigned short* __restrict__ attn,   // [M, D] bf16
         const unsigned short* __restrict__ wpt,    // [N, D] bf16 (Wp^T)
         const float* __restrict__ bp,
         float* __restrict__ out) {
    const int tid  = threadIdx.x;
    const int wave = tid >> 5, lane = tid & 31;
    const int grp  = lane >> 4, lr = lane & 15;
    const int wm   = wave >> 1, wn = wave & 1;
    const int m0   = blockIdx.x * 64 + wm * 16;
    const int n0   = blockIdx.y * 128 + wn * 64;

    const unsigned short* arow = attn + (size_t)(m0 + lr) * D_;
    const unsigned short* brow[4];
#pragma unroll
    for (int nt = 0; nt < 4; ++nt)
        brow[nt] = wpt + (size_t)(n0 + nt * 16 + lr) * D_;

    v8f c[4] = {v8f{}, v8f{}, v8f{}, v8f{}};
    for (int kb = 0; kb < D_; kb += 32) {
        v16bf a = load_frag(arow + kb, grp);
#pragma unroll
        for (int nt = 0; nt < 4; ++nt)
            c[nt] = wmma_bf16(a, load_frag(brow[nt] + kb, grp), c[nt]);
    }

#pragma unroll
    for (int nt = 0; nt < 4; ++nt) {
        const int   col  = n0 + nt * 16 + lr;
        const float bias = bp[col];
#pragma unroll
        for (int r = 0; r < 8; ++r)
            out[(size_t)(m0 + grp * 8 + r) * D_ + col] = c[nt][r] + bias;
    }
}

// ---------------------------------------------------------------------------
extern "C" void kernel_launch(void* const* d_in, const int* in_sizes, int n_in,
                              void* d_out, int out_size, void* d_ws, size_t ws_size,
                              hipStream_t stream) {
    const float* x  = (const float*)d_in[0];
    const float* Wq = (const float*)d_in[1];
    const float* Wk = (const float*)d_in[2];
    const float* Wv = (const float*)d_in[3];
    const float* Wp = (const float*)d_in[4];
    const float* bp = (const float*)d_in[5];
    float* out = (float*)d_out;

    // Workspace layout (ushort element offsets)
    unsigned short* ws  = (unsigned short*)d_ws;
    unsigned short* xb  = ws;                          // 8,388,608  (x bf16)
    unsigned short* wqt = ws + 8388608;                // 1,048,576  (Wq^T)
    unsigned short* wkt = wqt + 1048576;               // 1,048,576  (Wk^T)
    unsigned short* wvt = wkt + 1048576;               // 1,048,576  (Wv^T)
    unsigned short* wpt = wvt + 1048576;               // 1,048,576  (Wp^T)
    unsigned short* qbf = wpt + 1048576;               // 8,388,608  Q [BH,S,HD]
    unsigned short* kbf = qbf + 8388608;               // 8,388,608  K [BH,S,HD]
    unsigned short* vtb = kbf + 8388608;               // 8,388,608  V^T [BH,HD,S]
    unsigned short* atn = vtb + 8388608;               // 8,388,608  attn [B,S,D]

    cvt_x  <<<4096, 256, 0, stream>>>(x, xb, M_ * D_);
    cvt_w_t<<<1024, 256, 0, stream>>>(Wq, wqt);
    cvt_w_t<<<1024, 256, 0, stream>>>(Wk, wkt);
    cvt_w_t<<<1024, 256, 0, stream>>>(Wv, wvt);
    cvt_wp_t<<<1024, 256, 0, stream>>>(Wp, wpt);

    dim3 gp(H_, M_ / 64);
    gemm_proj<<<gp, 256, 0, stream>>>(xb, wqt, qbf, 0);
    gemm_proj<<<gp, 256, 0, stream>>>(xb, wkt, kbf, 0);
    gemm_proj<<<gp, 256, 0, stream>>>(xb, wvt, vtb, 1);

    flash_attn<<<dim3(B_ * H_, S_ / 64), 128, 0, stream>>>(qbf, kbf, vtb, atn);

    gemm_out<<<dim3(M_ / 64, D_ / 128), 256, 0, stream>>>(atn, wpt, bp, out);
}
